// InpaintGenerator_2688649528070
// MI455X (gfx1250) — matrix operand order repaired and checked by reading, not compiled, not run on hardware
//
#include <hip/hip_runtime.h>

// ---------------------------------------------------------------------------
// InpaintGenerator block for MI455X (gfx1250, wave32, WMMA, async-LDS).
//
// Shapes: B=8, C=64, H=W=256, heads=8, dk=8, PS=32 -> 64 windows of 32x32,
// windowed feature dim E = 8*32*32 = 8192.
//
// Kernels (all GEMMs on v_wmma_f32_16x16x32_f16):
//   1) repack:  conv weights f32 [O,I,3,3] -> f16 [tap][oc][ic] (L2-resident)
//   2) qkv:     conv1x1 -> windowed f16 Q,K,V [B,8,64,8192]
//   3) attn_qk: split-K (8 slices) S-partials, deterministic (no atomics)
//   4) softmax: reduce partials, softmax rows -> P f16 [bh][64][64]
//   5) attn_pv: Y = P V, parallel over 64 N-chunks, un-window to y16
//   6) conv3x3 x3: implicit GEMM over 9 taps, reflect pad, lrelu, residual
//
// Workspace (assumes ws_size >= 544 MiB):
//   qw 0, kw 64M, vw 128M, y16 192M, x1_f32 256M, x1_f16 384M, ff16 448M,
//   Spart 512M (8MB f32), Pbuf 521M (512KB f16), wrep 522M (216KB f16)
// ---------------------------------------------------------------------------

typedef __attribute__((ext_vector_type(16))) _Float16 v16h;
typedef __attribute__((ext_vector_type(8)))  float    v8f;

#define LANE (threadIdx.x & 31)
#define WAVE (threadIdx.x >> 5)

__device__ inline v8f wmma_f16(v16h a, v16h b, v8f c) {
  return __builtin_amdgcn_wmma_f32_16x16x32_f16(false, a, false, b, (short)0, c,
                                                false, false);
}

// CDNA5 async global->LDS copy of 16 bytes (GLOBAL_LOAD_ASYNC_TO_LDS_B128,
// GV addressing: vdst = LDS byte offset, vaddr = 64-bit global address).
__device__ inline void async_copy_b128(void* lds_dst, const void* gsrc) {
  unsigned lds = (unsigned)(size_t)lds_dst;           // low 32b = LDS offset
  unsigned long long ga = (unsigned long long)(size_t)gsrc;
  asm volatile("global_load_async_to_lds_b128 %0, %1, off"
               :: "v"(lds), "v"(ga) : "memory");
}
__device__ inline void async_wait0() {
  asm volatile("s_wait_asynccnt 0x0" ::: "memory");
}

// A fragment: 16x32 f16, row-major LDS tile, leading dim ld.
__device__ inline v16h frag_a(const _Float16* p, int ld, int row0, int k0) {
  int lane = LANE;
  int m = lane & 15;
  int koff = (lane < 16) ? 0 : 16;
  const _Float16* q = p + (size_t)(row0 + m) * ld + k0 + koff;
  v16h a;
#pragma unroll
  for (int i = 0; i < 16; ++i) a[i] = q[i];
  return a;
}

// B fragment: 32x16 f16 from row-major K x N tile (row stride ld).
__device__ inline v16h frag_b(const _Float16* p, int ld, int k0, int col0) {
  int lane = LANE;
  int n = lane & 15;
  int koff = (lane < 16) ? 0 : 16;
  const _Float16* q = p + (size_t)(k0 + koff) * ld + col0 + n;
  v16h b;
#pragma unroll
  for (int i = 0; i < 16; ++i) b[i] = q[(size_t)i * ld];
  return b;
}

// B fragment where the LDS tile is N x K row-major (B = tile^T).
__device__ inline v16h frag_bT(const _Float16* p, int ld, int k0, int col0) {
  int lane = LANE;
  int n = lane & 15;
  int koff = (lane < 16) ? 0 : 16;
  const _Float16* q = p + (size_t)(col0 + n) * ld + k0 + koff;
  v16h b;
#pragma unroll
  for (int i = 0; i < 16; ++i) b[i] = q[i];
  return b;
}

__device__ inline int refl(int i, int n) {
  if (i < 0) return -i;
  if (i >= n) return 2 * n - 2 - i;
  return i;
}

// ---------------------------------------------------------------------------
// Kernel: repack conv weights [O,I,3,3] f32 -> [tap][oc][ic] f16. grid=3.
// ---------------------------------------------------------------------------
__global__ void repack_w_kernel(const float* __restrict__ w0,
                                const float* __restrict__ w1,
                                const float* __restrict__ w2,
                                _Float16* __restrict__ wrep) {
  const int c = blockIdx.x;
  const float* src = (c == 0) ? w0 : (c == 1) ? w1 : w2;
  for (int idx = threadIdx.x; idx < 36864; idx += 256) {
    const int oc = idx / 576;
    const int ic = (idx / 9) % 64;
    const int tap = idx % 9;
    wrep[(size_t)c * 36864 + ((size_t)tap * 64 + oc) * 64 + ic] =
        (_Float16)src[idx];
  }
}

// ---------------------------------------------------------------------------
// Kernel: QKV conv1x1 -> windowed f16.  grid = B*H*(W/128) = 4096, 256 thr.
// ---------------------------------------------------------------------------
__global__ void qkv_kernel(const float* __restrict__ x,
                           const float* __restrict__ wq, const float* __restrict__ bq,
                           const float* __restrict__ wk, const float* __restrict__ bk,
                           const float* __restrict__ wv, const float* __restrict__ bv,
                           _Float16* __restrict__ qw, _Float16* __restrict__ kw,
                           _Float16* __restrict__ vw) {
  __shared__ _Float16 sW[3][64][64];   // 24 KB
  __shared__ _Float16 sX[64][128];     // 16 KB

  const int tid = threadIdx.x;
  const int bid = blockIdx.x;
  const int xseg = bid & 1;
  const int yy = (bid >> 1) & 255;
  const int b = bid >> 9;
  const int x0 = xseg * 128;

  __builtin_amdgcn_s_wait_tensorcnt(0);

  for (int idx = tid; idx < 3 * 4096; idx += 256) {
    int m = idx >> 12;
    int rem = idx & 4095;
    const float* wsrc = (m == 0) ? wq : (m == 1) ? wk : wv;
    sW[m][rem >> 6][rem & 63] = (_Float16)wsrc[rem];
  }
  {
    int ic = tid >> 2;
    int c0 = (tid & 3) * 32;
    const float* src = x + (((size_t)(b * 64 + ic) * 256 + yy) * 256 + x0 + c0);
#pragma unroll 8
    for (int i = 0; i < 32; ++i) sX[ic][c0 + i] = (_Float16)src[i];
  }
  __syncthreads();

  for (int t = WAVE; t < 96; t += 8) {
    const int rowtile = t >> 3;
    const int coltile = t & 7;
    const int m = rowtile >> 2;
    const int oc0 = (rowtile & 3) * 16;
    const int px0 = coltile * 16;

    v8f acc = {};
#pragma unroll
    for (int k0 = 0; k0 < 64; k0 += 32) {
      v16h a = frag_a(&sW[m][0][0], 64, oc0, k0);
      v16h bfr = frag_b(&sX[0][0], 128, k0, px0);
      acc = wmma_f16(a, bfr, acc);
    }

    const float* bias = (m == 0) ? bq : (m == 1) ? bk : bv;
    _Float16* dst = (m == 0) ? qw : (m == 1) ? kw : vw;
    const int lane = LANE;
    const int n = lane & 15;
    const int gx = x0 + px0 + n;
    const int win = ((yy >> 5) << 3) + (gx >> 5);
    const int py = yy & 31;
    const int pxw = gx & 31;
#pragma unroll
    for (int r = 0; r < 8; ++r) {
      const int M = (lane < 16) ? r : (r + 8);
      const int oc = oc0 + M;
      const int h = oc >> 3, dk = oc & 7;
      const int e = dk * 1024 + py * 32 + pxw;
      const size_t di = ((size_t)((b * 8 + h) * 64 + win)) * 8192 + e;
      dst[di] = (_Float16)(acc[r] + bias[oc]);
    }
  }
}

// ---------------------------------------------------------------------------
// Kernel: S-partials = Q K^T over one K-slice of 1024.
// grid = 64 bh * 8 slices = 512 blocks, 256 threads.
// Spart[bh][ks][64][64] f32 (deterministic split-K, summed in softmax).
// ---------------------------------------------------------------------------
__global__ void attn_qk_kernel(const _Float16* __restrict__ qw,
                               const _Float16* __restrict__ kw,
                               float* __restrict__ Spart) {
  __shared__ _Float16 sQ[64][128];  // 16 KB
  __shared__ _Float16 sK[64][128];  // 16 KB

  const int tid = threadIdx.x;
  const int bh = blockIdx.x >> 3;
  const int ks = blockIdx.x & 7;
  const size_t base = (size_t)bh * 64 * 8192;
  const int kbase = ks * 1024;
  const float scale = 0.011048543456039806f;  // 1/sqrt(8192)

  const int w = WAVE;
  const int row = tid >> 2;
  const int c0 = (tid & 3) * 32;
  v8f acc0 = {}, acc1 = {};

  for (int kc = 0; kc < 8; ++kc) {
    const int k0g = kbase + kc * 128;
    const _Float16* qs = qw + base + (size_t)row * 8192 + k0g + c0;
    const _Float16* ksrc = kw + base + (size_t)row * 8192 + k0g + c0;
#pragma unroll
    for (int i = 0; i < 4; ++i) {   // 32 f16 = 4x b128 per thread, per matrix
      async_copy_b128(&sQ[row][c0 + i * 8], qs + i * 8);
      async_copy_b128(&sK[row][c0 + i * 8], ksrc + i * 8);
    }
    if (kc + 1 < 8) {
      __builtin_prefetch(qs + 128, 0, 1);
      __builtin_prefetch(ksrc + 128, 0, 1);
    }
    async_wait0();
    __syncthreads();
#pragma unroll
    for (int k0 = 0; k0 < 128; k0 += 32) {
      {
        const int mt = w >> 2, nt = w & 3;
        v16h a = frag_a(&sQ[0][0], 128, mt * 16, k0);
        v16h bf = frag_bT(&sK[0][0], 128, k0, nt * 16);
        acc0 = wmma_f16(a, bf, acc0);
      }
      {
        const int t1 = w + 8;
        const int mt = t1 >> 2, nt = t1 & 3;
        v16h a = frag_a(&sQ[0][0], 128, mt * 16, k0);
        v16h bf = frag_bT(&sK[0][0], 128, k0, nt * 16);
        acc1 = wmma_f16(a, bf, acc1);
      }
    }
    __syncthreads();
  }

  float* out = Spart + ((size_t)(bh * 8 + ks)) * 4096;
  const int lane = LANE;
  const int n = lane & 15;
#pragma unroll
  for (int r = 0; r < 8; ++r) {
    const int M = (lane < 16) ? r : (r + 8);
    { const int mt = w >> 2, nt = w & 3;
      out[(mt * 16 + M) * 64 + nt * 16 + n] = acc0[r] * scale; }
    { const int t1 = w + 8; const int mt = t1 >> 2, nt = t1 & 3;
      out[(mt * 16 + M) * 64 + nt * 16 + n] = acc1[r] * scale; }
  }
}

// ---------------------------------------------------------------------------
// Kernel: reduce split-K partials + softmax.  grid = 64 bh, 64 threads.
// ---------------------------------------------------------------------------
__global__ void softmax_kernel(const float* __restrict__ Spart,
                               _Float16* __restrict__ Pbuf) {
  __shared__ float sS[64][65];
  const int bh = blockIdx.x;
  const int row = threadIdx.x;   // 0..63
  const float* sp = Spart + (size_t)bh * 8 * 4096;

  for (int j = 0; j < 64; ++j) {
    float v = 0.f;
#pragma unroll
    for (int ks = 0; ks < 8; ++ks) v += sp[ks * 4096 + row * 64 + j];
    sS[row][j] = v;
  }
  float mx = -3.0e38f;
#pragma unroll 8
  for (int j = 0; j < 64; ++j) mx = fmaxf(mx, sS[row][j]);
  float sum = 0.f;
#pragma unroll 8
  for (int j = 0; j < 64; ++j) sum += __expf(sS[row][j] - mx);
  const float inv = 1.f / sum;
  _Float16* pd = Pbuf + (size_t)bh * 4096 + row * 64;
#pragma unroll 8
  for (int j = 0; j < 64; ++j)
    pd[j] = (_Float16)(__expf(sS[row][j] - mx) * inv);
}

// ---------------------------------------------------------------------------
// Kernel: Y = P V for one 128-wide N chunk; un-window on store.
// grid = 64 bh * 64 chunks = 4096 blocks, 256 threads.
// ---------------------------------------------------------------------------
__global__ void attn_pv_kernel(const _Float16* __restrict__ Pbuf,
                               const _Float16* __restrict__ vw,
                               _Float16* __restrict__ y16) {
  __shared__ _Float16 sP[64][64];   // 8 KB
  __shared__ _Float16 sV[64][128];  // 16 KB

  const int tid = threadIdx.x;
  const int bh = blockIdx.x >> 6;
  const int nc = blockIdx.x & 63;
  const int b = bh >> 3, h = bh & 7;
  const size_t base = (size_t)bh * 64 * 8192;
  const int n0g = nc * 128;
  const int w = WAVE;
  const int row = tid >> 2;
  const int c0 = (tid & 3) * 32;

  // stage P (contiguous) and V chunk via async global->LDS
  {
    const _Float16* ps = Pbuf + (size_t)bh * 4096 + tid * 16;
    async_copy_b128(&sP[0][0] + tid * 16, ps);
    async_copy_b128(&sP[0][0] + tid * 16 + 8, ps + 8);
    const _Float16* vs = vw + base + (size_t)row * 8192 + n0g + c0;
#pragma unroll
    for (int i = 0; i < 4; ++i)
      async_copy_b128(&sV[row][c0 + i * 8], vs + i * 8);
  }
  async_wait0();
  __syncthreads();

#pragma unroll
  for (int i = 0; i < 4; ++i) {
    const int t = w * 4 + i;
    const int mt = t >> 3, ct = t & 7;
    v8f acc = {};
#pragma unroll
    for (int k0 = 0; k0 < 64; k0 += 32) {
      v16h a = frag_a(&sP[0][0], 64, mt * 16, k0);
      v16h bf = frag_b(&sV[0][0], 128, k0, ct * 16);
      acc = wmma_f16(a, bf, acc);
    }
    const int lane = LANE;
    const int n = lane & 15;
    const int e0 = n0g + ct * 16 + n;
    const int dk = e0 >> 10, py = (e0 >> 5) & 31, pxw = e0 & 31;
    const int ch = h * 8 + dk;
#pragma unroll
    for (int r = 0; r < 8; ++r) {
      const int M = (lane < 16) ? r : (r + 8);
      const int qwin = mt * 16 + M;
      const int gy = ((qwin >> 3) << 5) + py;
      const int gx = ((qwin & 7) << 5) + pxw;
      y16[((size_t)(b * 64 + ch) * 256 + gy) * 256 + gx] = (_Float16)acc[r];
    }
  }
}

// ---------------------------------------------------------------------------
// Kernel: conv3x3 (reflect, dilation d) + bias + lrelu + optional residual.
// 64-px strips: grid = B*H*(W/64) = 8192 blocks, 256 threads.
// out = (res ? res : 0) + lrelu(conv(in) + bias)
// ---------------------------------------------------------------------------
__global__ void conv3x3_kernel(const _Float16* __restrict__ in16,
                               const _Float16* __restrict__ wtap9,  // [9][64][64]
                               const float* __restrict__ bias,
                               const float* __restrict__ res32,
                               float* __restrict__ out32,
                               _Float16* __restrict__ out16,
                               int dil) {
  __shared__ _Float16 sIn[64 * 3 * 68];  // ~26 KB (ic, tap-row, halo col)
  __shared__ _Float16 sWt[64][64];       // 8 KB, one tap at a time

  const int tid = threadIdx.x;
  const int bid = blockIdx.x;
  const int seg = bid & 3;
  const int yy = (bid >> 2) & 255;
  const int b = bid >> 10;
  const int x0 = seg * 64;
  const int wide = 64 + 2 * dil;

  for (int idx = tid; idx < 64 * 3 * 68; idx += 256) {
    const int ic = idx / 204;
    const int r = (idx / 68) % 3;
    const int j = idx % 68;
    if (j < wide) {
      const int gy = refl(yy + (r - 1) * dil, 256);
      const int gx = refl(x0 + j - dil, 256);
      sIn[idx] = in16[((size_t)(b * 64 + ic) * 256 + gy) * 256 + gx];
    }
  }

  const int w = WAVE;
  v8f acc[2] = {{}, {}};

  for (int tap = 0; tap < 9; ++tap) {
    const int ky = tap / 3, kx = tap % 3;
    __syncthreads();   // prior tap's reads done (and sIn ready on tap 0)
    {  // stage this tap's 64x64 f16 weights (contiguous) via async copy
      const _Float16* wsrc = wtap9 + (size_t)tap * 4096 + tid * 16;
      async_copy_b128(&sWt[0][0] + tid * 16, wsrc);
      async_copy_b128(&sWt[0][0] + tid * 16 + 8, wsrc + 8);
    }
    async_wait0();
    __syncthreads();

#pragma unroll
    for (int i = 0; i < 2; ++i) {
      const int t = w + i * 8;
      const int oc0 = (t >> 2) * 16;
      const int px0 = (t & 3) * 16;
      const _Float16* bbase = sIn + ky * 68 + px0 + kx * dil;
#pragma unroll
      for (int k0 = 0; k0 < 64; k0 += 32) {
        v16h a = frag_a(&sWt[0][0], 64, oc0, k0);
        v16h bf = frag_b(bbase + (size_t)k0 * 204, 204, 0, 0);
        acc[i] = wmma_f16(a, bf, acc[i]);
      }
    }
  }

  const int lane = LANE;
  const int n = lane & 15;
#pragma unroll
  for (int i = 0; i < 2; ++i) {
    const int t = w + i * 8;
    const int oc0 = (t >> 2) * 16;
    const int px0 = (t & 3) * 16;
    const int gx = x0 + px0 + n;
#pragma unroll
    for (int r = 0; r < 8; ++r) {
      const int M = (lane < 16) ? r : (r + 8);
      const int oc = oc0 + M;
      float z = acc[i][r] + bias[oc];
      z = (z > 0.f) ? z : 0.2f * z;
      const size_t di = ((size_t)(b * 64 + oc) * 256 + yy) * 256 + gx;
      if (res32) z += res32[di];
      if (out32) out32[di] = z;
      if (out16) out16[di] = (_Float16)z;
    }
  }
}

// ---------------------------------------------------------------------------
extern "C" void kernel_launch(void* const* d_in, const int* in_sizes, int n_in,
                              void* d_out, int out_size, void* d_ws, size_t ws_size,
                              hipStream_t stream) {
  const float* x   = (const float*)d_in[0];
  // d_in[1] = mask m: unused (reference discards masked_fill result)
  const float* wq  = (const float*)d_in[2];
  const float* bq  = (const float*)d_in[3];
  const float* wk  = (const float*)d_in[4];
  const float* bk  = (const float*)d_in[5];
  const float* wv  = (const float*)d_in[6];
  const float* bv  = (const float*)d_in[7];
  const float* wo  = (const float*)d_in[8];
  const float* bo  = (const float*)d_in[9];
  const float* wf1 = (const float*)d_in[10];
  const float* bf1 = (const float*)d_in[11];
  const float* wf2 = (const float*)d_in[12];
  const float* bf2 = (const float*)d_in[13];

  char* ws = (char*)d_ws;
  const size_t MB = 1ull << 20;
  _Float16* qw    = (_Float16*)(ws + 0 * MB);
  _Float16* kw    = (_Float16*)(ws + 64 * MB);
  _Float16* vw    = (_Float16*)(ws + 128 * MB);
  _Float16* y16   = (_Float16*)(ws + 192 * MB);
  float*    x1f   = (float*)   (ws + 256 * MB);
  _Float16* x1h   = (_Float16*)(ws + 384 * MB);
  _Float16* ff16  = (_Float16*)(ws + 448 * MB);
  float*    Spart = (float*)   (ws + 512 * MB);  // 64*8*4096 f32 = 8 MB
  _Float16* Pbuf  = (_Float16*)(ws + 521 * MB);  // 64*4096 f16 = 512 KB
  _Float16* wrep  = (_Float16*)(ws + 522 * MB);  // 3*9*64*64 f16 = 216 KB

  repack_w_kernel<<<3, 256, 0, stream>>>(wo, wf1, wf2, wrep);
  qkv_kernel<<<4096, 256, 0, stream>>>(x, wq, bq, wk, bk, wv, bv, qw, kw, vw);
  attn_qk_kernel<<<512, 256, 0, stream>>>(qw, kw, Spart);
  softmax_kernel<<<64, 64, 0, stream>>>(Spart, Pbuf);
  attn_pv_kernel<<<4096, 256, 0, stream>>>(Pbuf, vw, y16);
  // x1 = x + lrelu(conv(y, wo))
  conv3x3_kernel<<<8192, 256, 0, stream>>>(y16, wrep, bo, x, x1f, x1h, 1);
  // ff = lrelu(conv(x1, wf1, dil=2))
  conv3x3_kernel<<<8192, 256, 0, stream>>>(x1h, wrep + 36864, bf1, nullptr,
                                           nullptr, ff16, 2);
  // out = x1 + lrelu(conv(ff, wf2))
  conv3x3_kernel<<<8192, 256, 0, stream>>>(ff16, wrep + 2 * 36864, bf2, x1f,
                                           (float*)d_out, nullptr, 1);
}
